// CNN_35716948033851
// MI455X (gfx1250) — matrix operand order repaired and checked by reading, not compiled, not run on hardware
//
#include <hip/hip_runtime.h>

// ---------------------------------------------------------------------------
// CNN sliding-window GEMM + max-over-time + ReLU for MI455X (gfx1250).
//   out[b,h] = relu(max_t( sum_f z[b,t,f] * W[h,f] ) + bias[h])
//   z[b,t, w*256+e] = input[b, t+w, e],  t in [0,508)
//
// Single-pass implicit GEMM: whole M=512 resident as 516 input rows in LDS
// (covers all 5 windows of K=1280), W streamed in double-buffered K-chunks.
// Per-wave 64x64 tile -> 16 ds_load_b128 per 16 v_wmma_f32_16x16x32_bf16.
// ---------------------------------------------------------------------------

namespace {
constexpr int kB     = 32;
constexpr int kWords = 512;
constexpr int kE     = 256;
constexpr int kWin   = 5;
constexpr int kF     = kWin * kE;         // 1280 (GEMM K)
constexpr int kH     = 1024;              // GEMM N
constexpr int kT     = kWords - kWin + 1; // 508 (GEMM M per batch)

constexpr int NT     = 64;                // h columns per workgroup
constexpr int MT     = 512;               // all t rows in one pass
constexpr int AROWS  = MT + kWin - 1;     // 516 staged input rows
constexpr int ASTR   = kE + 8;            // 264 bf16 (pad -> conflict-free b128)
constexpr int KC     = 128;               // W K-chunk
constexpr int NKC    = kF / KC;           // 10 chunks
constexpr int WCSTR  = KC + 8;            // 136 bf16
constexpr int ABYTES = AROWS * ASTR * 2;  // 272448
constexpr int WCBYTES = NT * WCSTR * 2;   // 17408
constexpr int SMEM   = ABYTES + 2 * WCBYTES; // 307264 <= 320 KB
constexpr float NEG_INF = -3.402823466e38f;
}

typedef __bf16 v16bf __attribute__((ext_vector_type(16)));
typedef float  v8f   __attribute__((ext_vector_type(8)));
typedef float  f32x4 __attribute__((ext_vector_type(4)));
typedef unsigned short u16x4 __attribute__((ext_vector_type(4)));

__device__ __forceinline__ unsigned short f2bf(float f) {
  union { float f; unsigned u; } c; c.f = f;
  unsigned u = c.u + 0x7FFFu + ((c.u >> 16) & 1u);   // round-to-nearest-even
  return (unsigned short)(u >> 16);
}

__global__ void __launch_bounds__(256)
cnn_win_gemm_max(const float* __restrict__ inp,   // [32][512][256]
                 const float* __restrict__ Wg,    // [1024][1280]
                 const float* __restrict__ bias,  // [1024]
                 float* __restrict__ out) {       // [32][1024]
  extern __shared__ unsigned short smem[];
  unsigned short* lds_a = smem;                       // [AROWS][ASTR] bf16
  unsigned short* wbuf0 = smem + ABYTES / 2;          // [NT][WCSTR]  bf16
  unsigned short* wbuf1 = wbuf0 + WCBYTES / 2;

  const int tid  = threadIdx.x;
  const int lane = tid & 31;
  const int wave = tid >> 5;    // wave w owns rows [64w, 64w+64)
  const int half = lane >> 4;   // wave32 lane-half (WMMA operand layout)
  const int ln   = lane & 15;

  const int h0 = blockIdx.x * NT;
  const int b  = blockIdx.y;

  // ---- Stage all 516 input rows once (f32 -> bf16); covers all 5 windows --
  for (int idx = tid; idx < AROWS * (kE / 4); idx += 256) {
    const int row = idx >> 6, c4 = idx & 63;
    int g = row; if (g > kWords - 1) g = kWords - 1;   // clamp (masked later)
    f32x4 v = ((const f32x4*)(inp + ((size_t)b * kWords + g) * kE))[c4];
    u16x4 p = { f2bf(v.x), f2bf(v.y), f2bf(v.z), f2bf(v.w) };
    *(u16x4*)(lds_a + row * ASTR + c4 * 4) = p;
  }

  // ---- W K-chunk stage helpers (register-buffered double pipeline) --------
  // One chunk: 64 rows x 128 k = 2048 f32x4, 8 per thread. tid-major within
  // a row (32 f32x4 per row) -> fully coalesced 512B wave reads.
  f32x4 wreg[8];
  const int wrow = 0;  (void)wrow;
  auto loadW = [&](int kc) {
    const int kbase = kc * (KC / 4);
    #pragma unroll
    for (int i = 0; i < 8; ++i) {
      const int idx = tid + i * 256;
      const int row = idx >> 5, c4 = idx & 31;
      wreg[i] = ((const f32x4*)(Wg + (size_t)(h0 + row) * kF))[kbase + c4];
    }
  };
  auto storeW = [&](unsigned short* buf) {
    #pragma unroll
    for (int i = 0; i < 8; ++i) {
      const int idx = tid + i * 256;
      const int row = idx >> 5, c4 = idx & 31;
      u16x4 p = { f2bf(wreg[i].x), f2bf(wreg[i].y),
                  f2bf(wreg[i].z), f2bf(wreg[i].w) };
      *(u16x4*)(buf + row * WCSTR + c4 * 4) = p;
    }
  };

  loadW(0);
  storeW(wbuf0);
  __syncthreads();   // A + first W chunk visible

  const int half8  = half * 8;
  const int half16 = half * 16;
  const int rowA   = wave * 64 + ln;   // A-frag row base (M = lane&15 per ISA)

  v8f acc[4][4] = {};   // [ms][ns] 16x16 tiles: 64 rows x 64 cols per wave

  for (int kc = 0; kc < NKC; ++kc) {
    unsigned short* cur = (kc & 1) ? wbuf1 : wbuf0;
    unsigned short* nxt = (kc & 1) ? wbuf0 : wbuf1;
    if (kc + 1 < NKC) loadW(kc + 1);   // global loads overlap 64 WMMAs below

    const int w  = kc >> 1;                 // window index (row offset in A)
    const int e0base = (kc & 1) * 128;      // embedding col base of chunk
    #pragma unroll
    for (int ks = 0; ks < 4; ++ks) {
      const int e0   = e0base + ks * 32;
      const int koff = ks * 32;
      union Frag { f32x4 q[2]; v16bf v; };
      Frag a[4], bb[4];
      #pragma unroll
      for (int ms = 0; ms < 4; ++ms) {
        // A 16x32 bf16 layout: lane<16 -> K{0..7}|{16..23}, lane>=16 -> +8
        const unsigned short* pa =
            lds_a + (rowA + ms * 16 + w) * ASTR + e0 + half8;
        a[ms].q[0] = *(const f32x4*)pa;          // ds_load_b128
        a[ms].q[1] = *(const f32x4*)(pa + 16);   // ds_load_b128
      }
      #pragma unroll
      for (int ns = 0; ns < 4; ++ns) {
        // B 32x16 bf16 layout: lane<16 -> K{0..7}|{16..23}, lane>=16 -> +8
        const unsigned short* pb =
            cur + (ns * 16 + ln) * WCSTR + koff + half8;
        bb[ns].q[0] = *(const f32x4*)pb;
        bb[ns].q[1] = *(const f32x4*)(pb + 16);
      }
      #pragma unroll
      for (int ms = 0; ms < 4; ++ms)
        #pragma unroll
        for (int ns = 0; ns < 4; ++ns)
          acc[ms][ns] = __builtin_amdgcn_wmma_f32_16x16x32_bf16(
              false, a[ms].v, false, bb[ns].v, (short)0, acc[ms][ns],
              false, false);
    }

    if (kc + 1 < NKC) storeW(nxt);     // nxt's readers finished last barrier
    __syncthreads();
  }

  // ---- Max over time: tile rows, lane halves, waves; bias + ReLU ----------
  float* red = (float*)wbuf0;   // [8 waves][4 colgroup][16]; safe after barrier
  #pragma unroll
  for (int ns = 0; ns < 4; ++ns) {
    float v = NEG_INF;
    #pragma unroll
    for (int ms = 0; ms < 4; ++ms) {
      const int mBase = wave * 64 + ms * 16 + half * 8;  // + r = global t
      #pragma unroll
      for (int r = 0; r < 8; ++r) {
        float x = acc[ms][ns][r];
        if (mBase + r >= kT) x = NEG_INF;                // mask t in [508,512)
        v = fmaxf(v, x);
      }
    }
    v = fmaxf(v, __shfl_xor(v, 16, 32));  // lane l and l+16 share column N
    if (lane < 16) red[(wave * 4 + ns) * 16 + ln] = v;
  }
  __syncthreads();
  if (tid < NT) {
    const int ns = tid >> 4, l = tid & 15;
    float v = red[ns * 16 + l];
    #pragma unroll
    for (int wv = 1; wv < 8; ++wv) v = fmaxf(v, red[(wv * 4 + ns) * 16 + l]);
    v += bias[h0 + tid];
    out[(size_t)b * kH + h0 + tid] = fmaxf(v, 0.0f);
  }
}

extern "C" void kernel_launch(void* const* d_in, const int* in_sizes, int n_in,
                              void* d_out, int out_size, void* d_ws, size_t ws_size,
                              hipStream_t stream) {
  (void)in_sizes; (void)n_in; (void)out_size; (void)d_ws; (void)ws_size;
  const float* inp  = (const float*)d_in[0];
  const float* Wg   = (const float*)d_in[1];
  const float* bias = (const float*)d_in[2];
  float* out = (float*)d_out;

  static_assert(SMEM <= 320 * 1024, "LDS budget exceeded");
  hipFuncSetAttribute((const void*)cnn_win_gemm_max,
                      hipFuncAttributeMaxDynamicSharedMemorySize, SMEM);
  dim3 grid(kH / NT, kB);  // (16 h-tiles, 32 batches) = 512 workgroups
  cnn_win_gemm_max<<<grid, 256, SMEM, stream>>>(inp, Wg, bias, out);
}
